// VisionMambaBlock_82781199663292
// MI455X (gfx1250) — matrix operand order repaired and checked by reading, not compiled, not run on hardware
//
#include <hip/hip_runtime.h>
#include <hip/hip_bf16.h>
#include <math.h>

// ---------------------------------------------------------------------------
// VisionMambaBlock on gfx1250 (MI455X): WMMA f16->f32 GEMMs with LDS-staged
// B tiles + LDS-broadcast sequential selective scan.
// ---------------------------------------------------------------------------

#define D_MODEL   256
#define D_INNER   512
#define D_STATE   16
#define DT_RANK   16
#define DBL_DIM   48          // DT_RANK + 2*D_STATE
#define BATCH     4
#define LSEQ      4096        // 64*64
#define MROWS     (BATCH * LSEQ)   // 16384

typedef __attribute__((ext_vector_type(16))) _Float16 v16h;
typedef __attribute__((ext_vector_type(8)))  _Float16 v8h;
typedef __attribute__((ext_vector_type(8)))  float    v8f;

// ---------------------------------------------------------------------------
// A-fragment loader (contiguous k order, same as the LDS B fragment)
//   A (16x32, MxK): lane m=lane&15, h=lane>>4 holds K = 16h..16h+15
// ---------------------------------------------------------------------------
__device__ inline v16h load_a_frag_f16(const _Float16* __restrict__ A,
                                       size_t row, int K, int kk, int h) {
  const v8h* p0 = reinterpret_cast<const v8h*>(A + row * (size_t)K + kk + 16 * h);
  const v8h* p1 = reinterpret_cast<const v8h*>(A + row * (size_t)K + kk + 16 * h + 8);
  v8h c0 = *p0, c1 = *p1;
  v16h r;
#pragma unroll
  for (int i = 0; i < 8; ++i) { r[i] = c0[i]; r[8 + i] = c1[i]; }
  return r;
}

// ---------------------------------------------------------------------------
// Generic GEMM: C[M,N] = A[M,K] * W[N,K]^T    (f16 in, f32 accum, OutT out)
// Block = 8 waves (256 thr); each wave computes one 16 x (16*NTILES) strip.
// The W tile (shared by all 8 waves) is staged in LDS once per K-step:
//   B (32x16, KxN) fragment: lane n=lane&15, h=lane>>4 holds K = 16h..16h+15,
//   and B[k][n] = W[n][k], so each lane does one contiguous 32B ds read.
// rev!=0 un-reverses rows per batch (backward-direction output GEMM).
// ---------------------------------------------------------------------------
template <typename OutT, int NTILES>
__global__ void gemm_wmma_kernel(const _Float16* __restrict__ A,
                                 const _Float16* __restrict__ W,
                                 OutT* __restrict__ C,
                                 int M, int N, int K, int Ls, int rev) {
  __shared__ __align__(32) _Float16 sW[NTILES * 16 * 32];

  const int tid  = threadIdx.x;
  const int wave = tid >> 5;
  const int lane = tid & 31;
  const int h    = lane >> 4;
  const int m0   = (blockIdx.x * 8 + wave) * 16;
  const int n0   = blockIdx.y * (16 * NTILES);

  v8f acc[NTILES];
#pragma unroll
  for (int t = 0; t < NTILES; ++t) acc[t] = (v8f){0.f,0.f,0.f,0.f,0.f,0.f,0.f,0.f};

  const size_t arow = (size_t)(m0 + (lane & 15));
  const int wr = tid >> 2, wc = tid & 3;     // cooperative W-tile loader coords

  for (int kk = 0; kk < K; kk += 32) {
    // ---- stage W[n0 .. n0+16*NTILES)[kk .. kk+32) into LDS ----------------
    if (tid < NTILES * 64)
      *reinterpret_cast<v8h*>(&sW[wr * 32 + wc * 8]) =
          *reinterpret_cast<const v8h*>(W + (size_t)(n0 + wr) * K + kk + wc * 8);
    __syncthreads();

    if (kk + 32 < K)
      __builtin_prefetch(A + arow * (size_t)K + kk + 32, 0, 1);

    v16h af = load_a_frag_f16(A, arow, K, kk, h);
#pragma unroll
    for (int t = 0; t < NTILES; ++t) {
      v16h bf = *reinterpret_cast<const v16h*>(&sW[(t * 16 + (lane & 15)) * 32 + 16 * h]);
      acc[t] = __builtin_amdgcn_wmma_f32_16x16x32_f16(
          /*neg_a=*/false, af, /*neg_b=*/false, bf,
          /*c_mod=*/(short)0, acc[t], /*reuse_a=*/false, /*reuse_b=*/false);
    }
    __syncthreads();   // all frag reads done before next K-step overwrites sW
  }

  // C/D layout: VGPR v, lanes 0-15 -> (M=v, N=lane); lanes 16-31 -> (M=8+v)
#pragma unroll
  for (int t = 0; t < NTILES; ++t) {
    const int ncol = n0 + t * 16 + (lane & 15);
#pragma unroll
    for (int v = 0; v < 8; ++v) {
      int m = m0 + v + 8 * h;
      int dr = rev ? ((m / Ls) * Ls + (Ls - 1 - (m % Ls))) : m;
      C[(size_t)dr * N + ncol] = (OutT)acc[t][v];
    }
  }
}

// ---------------------------------------------------------------------------
// f32 -> f16 weight conversion
// ---------------------------------------------------------------------------
__global__ void cvt_f16_kernel(const float* __restrict__ s, _Float16* __restrict__ d, int n) {
  int i = blockIdx.x * blockDim.x + threadIdx.x;
  if (i < n) d[i] = (_Float16)s[i];
}

__device__ inline float silu_f(float x) { return x / (1.f + __expf(-x)); }

// ---------------------------------------------------------------------------
// style = silu(time_emb) @ ada_w^T + ada_b ; split into scale/shift
// ---------------------------------------------------------------------------
__global__ void style_kernel(const float* __restrict__ temb,
                             const float* __restrict__ ada_w,
                             const float* __restrict__ ada_b,
                             float* __restrict__ scale, float* __restrict__ shift) {
  int b = blockIdx.x, e = threadIdx.x;           // e in [0,512)
  float s = ada_b[e];
  for (int d = 0; d < D_MODEL; ++d)
    s += silu_f(temb[b * D_MODEL + d]) * ada_w[e * D_MODEL + d];
  if (e < D_MODEL) scale[b * D_MODEL + e] = s;
  else             shift[b * D_MODEL + (e - D_MODEL)] = s;
}

// ---------------------------------------------------------------------------
// LayerNorm + adaLN modulation. One wave per (b,l) row; writes forward and
// per-batch reversed f16 copies (so both mamba directions share kernels).
// ---------------------------------------------------------------------------
__global__ void ln_mod_kernel(const float* __restrict__ x,
                              const float* __restrict__ lng, const float* __restrict__ lnb,
                              const float* __restrict__ scale, const float* __restrict__ shift,
                              _Float16* __restrict__ xn, _Float16* __restrict__ xnr) {
  const int row  = blockIdx.x * 8 + (threadIdx.x >> 5);
  const int lane = threadIdx.x & 31;
  const int b = row >> 12, l = row & (LSEQ - 1);
  const float* xb = x + (size_t)b * D_MODEL * LSEQ;

  float vals[8], sum = 0.f, sq = 0.f;
#pragma unroll
  for (int i = 0; i < 8; ++i) {
    int c = lane + 32 * i;
    float v = xb[(size_t)c * LSEQ + l];
    vals[i] = v; sum += v; sq += v * v;
  }
#pragma unroll
  for (int off = 16; off; off >>= 1) {
    sum += __shfl_xor(sum, off, 32);
    sq  += __shfl_xor(sq,  off, 32);
  }
  const float mu  = sum * (1.f / D_MODEL);
  const float var = sq * (1.f / D_MODEL) - mu * mu;
  const float inv = rsqrtf(var + 1e-5f);

  const size_t ro = (size_t)row * D_MODEL;
  const size_t rr = ((size_t)b * LSEQ + (LSEQ - 1 - l)) * D_MODEL;
#pragma unroll
  for (int i = 0; i < 8; ++i) {
    int c = lane + 32 * i;
    float v = (vals[i] - mu) * inv * lng[c] + lnb[c];
    v = v * (1.f + scale[b * D_MODEL + c]) + shift[b * D_MODEL + c];
    xn[ro + c]  = (_Float16)v;
    xnr[rr + c] = (_Float16)v;
  }
}

// ---------------------------------------------------------------------------
// Causal depthwise conv (k=4) + SiLU. xi = xz[:, :512] (f16). Emits f32 (scan)
// and f16 (xproj GEMM A) copies.
// ---------------------------------------------------------------------------
__global__ void conv_silu_kernel(const _Float16* __restrict__ xz,
                                 const float* __restrict__ convW,
                                 const float* __restrict__ convB,
                                 float* __restrict__ xcf, _Float16* __restrict__ xch) {
  const int idx = blockIdx.x * blockDim.x + threadIdx.x;  // over MROWS*512
  const int d = idx & (D_INNER - 1);
  const int row = idx >> 9;
  const int l = row & (LSEQ - 1);
  float s = convB[d];
#pragma unroll
  for (int j = 0; j < 4; ++j) {
    int ls = l - 3 + j;
    if (ls >= 0)
      s += convW[d * 4 + j] * (float)xz[(size_t)(row - 3 + j) * (2 * D_INNER) + d];
  }
  float v = silu_f(s);
  xcf[(size_t)row * D_INNER + d] = v;
  xch[(size_t)row * D_INNER + d] = (_Float16)v;
}

// ---------------------------------------------------------------------------
// Selective scan: one thread per (b,d) chain, 16 h-states in VGPRs; the
// per-timestep (dt_r | B | C) row is staged in LDS once per block.
// Produces gated output y_gated in f16 for the outW GEMM.
// ---------------------------------------------------------------------------
__global__ void scan_kernel(const float* __restrict__ dbl,
                            const _Float16* __restrict__ xz,
                            const float* __restrict__ xcf,
                            const float* __restrict__ dtW,
                            const float* __restrict__ dtB,
                            const float* __restrict__ Alog,
                            const float* __restrict__ Dp,
                            _Float16* __restrict__ yg) {
  __shared__ float sdbl[DBL_DIM];
  const int b = blockIdx.x >> 1;
  const int d = ((blockIdx.x & 1) << 8) + threadIdx.x;   // 256 threads/block

  float Ar[D_STATE], Wr[D_STATE], h[D_STATE];
#pragma unroll
  for (int n = 0; n < D_STATE; ++n) {
    Ar[n] = -__expf(Alog[d * D_STATE + n]);
    Wr[n] = dtW[d * DT_RANK + n];
    h[n]  = 0.f;
  }
  const float dtb = dtB[d], dpv = Dp[d];

  for (int l = 0; l < LSEQ; ++l) {
    const size_t row = (size_t)b * LSEQ + l;
    __syncthreads();
    if (threadIdx.x < DBL_DIM) sdbl[threadIdx.x] = dbl[row * DBL_DIM + threadIdx.x];
    __syncthreads();

    float dt = dtb;
#pragma unroll
    for (int r = 0; r < DT_RANK; ++r) dt += Wr[r] * sdbl[r];
    dt = (dt > 20.f) ? dt : log1pf(__expf(dt));          // softplus

    const float xt  = xcf[row * D_INNER + d];
    const float dtx = dt * xt;
    float y = 0.f;
#pragma unroll
    for (int n = 0; n < D_STATE; ++n) {
      float dA = __expf(dt * Ar[n]);
      h[n] = dA * h[n] + dtx * sdbl[DT_RANK + n];
      y += h[n] * sdbl[DT_RANK + D_STATE + n];
    }
    const float zv = (float)xz[row * (2 * D_INNER) + D_INNER + d];
    yg[row * D_INNER + d] = (_Float16)((y + xt * dpv) * silu_f(zv));
  }
}

// ---------------------------------------------------------------------------
// sum + convert for the final projection A-matrix
// ---------------------------------------------------------------------------
__global__ void sum_cvt_kernel(const float* __restrict__ a, const float* __restrict__ b,
                               _Float16* __restrict__ d, int n) {
  int i = blockIdx.x * blockDim.x + threadIdx.x;
  if (i < n) d[i] = (_Float16)(a[i] + b[i]);
}

// ---------------------------------------------------------------------------
// residual + bias, back to (B,C,H,W)
// ---------------------------------------------------------------------------
__global__ void final_kernel(const float* __restrict__ x,
                             const float* __restrict__ pout,
                             const float* __restrict__ pb,
                             float* __restrict__ out) {
  int i = blockIdx.x * blockDim.x + threadIdx.x;    // over B*C*L
  int l = i & (LSEQ - 1);
  int c = (i >> 12) & (D_MODEL - 1);
  int b = i >> 20;
  out[i] = x[i] + pout[((size_t)b * LSEQ + l) * D_MODEL + c] + pb[c];
}

// ---------------------------------------------------------------------------
// Host launcher
// ---------------------------------------------------------------------------
extern "C" void kernel_launch(void* const* d_in, const int* in_sizes, int n_in,
                              void* d_out, int out_size, void* d_ws, size_t ws_size,
                              hipStream_t stream) {
  (void)in_sizes; (void)n_in; (void)out_size; (void)ws_size;

  const float* x      = (const float*)d_in[0];
  const float* temb   = (const float*)d_in[1];
  const float* ln_g   = (const float*)d_in[2];
  const float* ln_b   = (const float*)d_in[3];
  const float* ada_w  = (const float*)d_in[4];
  const float* ada_b  = (const float*)d_in[5];
  const float* proj_w = (const float*)d_in[6];
  const float* proj_b = (const float*)d_in[7];
  // per-direction params: inW, convW, convB, xprojW, dtW, dtB, Alog, D, outW
  const float* P[2][9];
  for (int dir = 0; dir < 2; ++dir)
    for (int k = 0; k < 9; ++k)
      P[dir][k] = (const float*)d_in[8 + dir * 9 + k];

  char* ws = (char*)d_ws;
  size_t off = 0;
  auto alloc = [&](size_t bytes) { size_t o = off; off = (off + bytes + 255) & ~(size_t)255; return o; };

  const size_t M = MROWS;
  _Float16* xn   = (_Float16*)(ws + alloc(M * D_MODEL * 2));
  _Float16* xnr  = (_Float16*)(ws + alloc(M * D_MODEL * 2));
  _Float16* w_in[2], *w_xp[2], *w_out[2];
  for (int dir = 0; dir < 2; ++dir) {
    w_in[dir]  = (_Float16*)(ws + alloc((size_t)2 * D_INNER * D_MODEL * 2));
    w_xp[dir]  = (_Float16*)(ws + alloc((size_t)DBL_DIM * D_INNER * 2));
    w_out[dir] = (_Float16*)(ws + alloc((size_t)D_MODEL * D_INNER * 2));
  }
  _Float16* w_pj = (_Float16*)(ws + alloc((size_t)D_MODEL * D_MODEL * 2));
  float* scaleb  = (float*)(ws + alloc((size_t)BATCH * D_MODEL * 4));
  float* shiftb  = (float*)(ws + alloc((size_t)BATCH * D_MODEL * 4));
  // per-direction scratch (reused; directions run sequentially on `stream`)
  _Float16* xz   = (_Float16*)(ws + alloc(M * 2 * D_INNER * 2));
  float*    xcf  = (float*)   (ws + alloc(M * D_INNER * 4));
  _Float16* xch  = (_Float16*)(ws + alloc(M * D_INNER * 2));
  float*    dbl  = (float*)   (ws + alloc(M * DBL_DIM * 4));
  _Float16* yg   = (_Float16*)(ws + alloc(M * D_INNER * 2));
  float* odir[2];
  odir[0] = (float*)(ws + alloc(M * D_MODEL * 4));
  odir[1] = (float*)(ws + alloc(M * D_MODEL * 4));
  _Float16* sumh = (_Float16*)(ws + alloc(M * D_MODEL * 2));
  float*    pout = (float*)   (ws + alloc(M * D_MODEL * 4));

  // ---- weight conversion -------------------------------------------------
  auto cvt = [&](const float* s, _Float16* d, int n) {
    cvt_f16_kernel<<<(n + 255) / 256, 256, 0, stream>>>(s, d, n);
  };
  for (int dir = 0; dir < 2; ++dir) {
    cvt(P[dir][0], w_in[dir],  2 * D_INNER * D_MODEL);
    cvt(P[dir][3], w_xp[dir],  DBL_DIM * D_INNER);
    cvt(P[dir][8], w_out[dir], D_MODEL * D_INNER);
  }
  cvt(proj_w, w_pj, D_MODEL * D_MODEL);

  // ---- adaLN style + layernorm/modulation --------------------------------
  style_kernel<<<BATCH, 2 * D_MODEL, 0, stream>>>(temb, ada_w, ada_b, scaleb, shiftb);
  ln_mod_kernel<<<M / 8, 256, 0, stream>>>(x, ln_g, ln_b, scaleb, shiftb, xn, xnr);

  // ---- two mamba directions (shared scratch, serialized on stream) -------
  for (int dir = 0; dir < 2; ++dir) {
    const _Float16* a_in = dir ? xnr : xn;
    // xz = x_norm @ inW^T   (16384 x 256 -> 16384 x 1024)
    gemm_wmma_kernel<_Float16, 4><<<dim3(M / 128, (2 * D_INNER) / 64), 256, 0, stream>>>(
        a_in, w_in[dir], xz, (int)M, 2 * D_INNER, D_MODEL, LSEQ, 0);
    // causal depthwise conv + silu
    conv_silu_kernel<<<(M * D_INNER) / 256, 256, 0, stream>>>(
        xz, P[dir][1], P[dir][2], xcf, xch);
    // dbl = xc @ xprojW^T   (16384 x 512 -> 16384 x 48)
    gemm_wmma_kernel<float, 3><<<dim3(M / 128, 1), 256, 0, stream>>>(
        xch, w_xp[dir], dbl, (int)M, DBL_DIM, D_INNER, LSEQ, 0);
    // selective scan + gating
    scan_kernel<<<BATCH * 2, 256, 0, stream>>>(
        dbl, xz, xcf, P[dir][4], P[dir][5], P[dir][6], P[dir][7], yg);
    // out_dir = y_gated @ outW^T (un-reverse rows for the backward pass)
    gemm_wmma_kernel<float, 4><<<dim3(M / 128, D_MODEL / 64), 256, 0, stream>>>(
        yg, w_out[dir], odir[dir], (int)M, D_MODEL, D_INNER, LSEQ, dir);
  }

  // ---- combine, project, residual ----------------------------------------
  sum_cvt_kernel<<<(M * D_MODEL) / 256, 256, 0, stream>>>(
      odir[0], odir[1], sumh, (int)(M * D_MODEL));
  gemm_wmma_kernel<float, 4><<<dim3(M / 128, D_MODEL / 64), 256, 0, stream>>>(
      sumh, w_pj, pout, (int)M, D_MODEL, D_MODEL, LSEQ, 0);
  final_kernel<<<(M * D_MODEL) / 256, 256, 0, stream>>>(x, pout, proj_b, (float*)d_out);
}
